// TORCHIEKF_66675072303163
// MI455X (gfx1250) — matrix operand order repaired and checked by reading, not compiled, not run on hardware
//
#include <hip/hip_runtime.h>
#include <math.h>

#define P_DIM 21
#define LD    33          // 32x32 tiles padded to stride 33 (bank-conflict avoidance)
#define EPSF  1e-8f
#define GACC  9.80655f

typedef float    v2f  __attribute__((ext_vector_type(2)));
typedef float    v8f  __attribute__((ext_vector_type(8)));
typedef _Float16 v16h __attribute__((ext_vector_type(16)));

// ======================= per-lane replicated 3x3 algebra =======================
__device__ __forceinline__ void skew3(const float* x, float* S) {
  S[0] = 0.f;   S[1] = -x[2]; S[2] = x[1];
  S[3] = x[2];  S[4] = 0.f;   S[5] = -x[0];
  S[6] = -x[1]; S[7] = x[0];  S[8] = 0.f;
}
__device__ __forceinline__ void m3mul(const float* A, const float* B, float* C) {
#pragma unroll
  for (int r = 0; r < 3; ++r)
#pragma unroll
    for (int c = 0; c < 3; ++c)
      C[r*3+c] = A[r*3+0]*B[0+c] + A[r*3+1]*B[3+c] + A[r*3+2]*B[6+c];
}
__device__ __forceinline__ void m3tmulm(const float* A, const float* B, float* C) { // A^T * B
#pragma unroll
  for (int r = 0; r < 3; ++r)
#pragma unroll
    for (int c = 0; c < 3; ++c)
      C[r*3+c] = A[0+r]*B[0+c] + A[3+r]*B[3+c] + A[6+r]*B[6+c];
}
__device__ __forceinline__ void m3vec(const float* A, const float* x, float* y) {
#pragma unroll
  for (int r = 0; r < 3; ++r)
    y[r] = A[r*3+0]*x[0] + A[r*3+1]*x[1] + A[r*3+2]*x[2];
}
__device__ __forceinline__ void m3tvec(const float* A, const float* x, float* y) {
#pragma unroll
  for (int r = 0; r < 3; ++r)
    y[r] = A[0+r]*x[0] + A[3+r]*x[1] + A[6+r]*x[2];
}
// branch-free row extract of a register-resident 3x3 (r in 0..2)
__device__ __forceinline__ void row3(const float* M, int r, float* o) {
  o[0] = (r==0) ? M[0] : ((r==1) ? M[3] : M[6]);
  o[1] = (r==0) ? M[1] : ((r==1) ? M[4] : M[7]);
  o[2] = (r==0) ? M[2] : ((r==1) ? M[5] : M[8]);
}
__device__ __forceinline__ void so3exp(const float* phi, float* R) {
  float ang = sqrtf(phi[0]*phi[0] + phi[1]*phi[1] + phi[2]*phi[2]);
  float sk[9];
  if (ang < EPSF) {
    skew3(phi, sk);
#pragma unroll
    for (int i = 0; i < 9; ++i) R[i] = sk[i];
    R[0] += 1.f; R[4] += 1.f; R[8] += 1.f;
  } else {
    float ax[3] = {phi[0]/ang, phi[1]/ang, phi[2]/ang};
    float s = sinf(ang), c = cosf(ang);
    skew3(ax, sk);
#pragma unroll
    for (int r = 0; r < 3; ++r)
#pragma unroll
      for (int cc = 0; cc < 3; ++cc)
        R[r*3+cc] = ((r==cc)?c:0.f) + (1.f-c)*ax[r]*ax[cc] + s*sk[r*3+cc];
  }
}
__device__ __forceinline__ void sen3exp_rj(const float* phi, float* R, float* J) {
  float ang = sqrtf(phi[0]*phi[0] + phi[1]*phi[1] + phi[2]*phi[2]);
  float sk[9];
  if (ang < EPSF) {
    skew3(phi, sk);
#pragma unroll
    for (int i = 0; i < 9; ++i) { R[i] = sk[i]; J[i] = 0.5f*sk[i]; }
    R[0] += 1.f; R[4] += 1.f; R[8] += 1.f;
    J[0] += 1.f; J[4] += 1.f; J[8] += 1.f;
  } else {
    float ax[3] = {phi[0]/ang, phi[1]/ang, phi[2]/ang};
    float s = sinf(ang), c = cosf(ang);
    float sa = s / ang, omc = (1.f - c) / ang;
    skew3(ax, sk);
#pragma unroll
    for (int r = 0; r < 3; ++r)
#pragma unroll
      for (int cc = 0; cc < 3; ++cc) {
        float o = ax[r]*ax[cc];
        R[r*3+cc] = ((r==cc)?c :0.f) + (1.f-c )*o + s  *sk[r*3+cc];
        J[r*3+cc] = ((r==cc)?sa:0.f) + (1.f-sa)*o + omc*sk[r*3+cc];
      }
  }
}
__device__ __forceinline__ void from_rpy(float r, float p, float y, float* R) {
  float cr=cosf(r), sr=sinf(r), cp=cosf(p), sp=sinf(p), cy=cosf(y), sy=sinf(y);
  R[0]=cy*cp; R[1]=cy*sp*sr - sy*cr; R[2]=cy*sp*cr + sy*sr;
  R[3]=sy*cp; R[4]=sy*sp*sr + cy*cr; R[5]=sy*sp*cr - cy*sr;
  R[6]=-sp;   R[7]=cp*sr;            R[8]=cp*cr;
}

// ============ structure-aware padded GEMM on one wave via WMMA f32 16x16x4 ============
// C = A * B (or A * B^T when TB). Operands stride-LD in LDS.
// ROWS/COLS: number of 16-wide output tiles; KS: number of K=4 WMMA steps.
// Row-tile loop kept rolled to bound live VGPRs; A fragment reused across column tiles.
template <bool TB, int ROWS, int COLS, int KS>
__device__ __forceinline__ void mm32(const float* A, const float* B, float* C, int lane) {
  const int lm = lane & 15;
  const int hi = lane >> 4;
#if defined(__gfx1250__) && __has_builtin(__builtin_amdgcn_wmma_f32_16x16x4_f32)
  const int kh = hi * 2;
#pragma unroll 1
  for (int ti = 0; ti < ROWS * 16; ti += 16) {
    v8f acc0 = {}, acc1 = {};
#pragma unroll
    for (int ks = 0; ks < KS; ++ks) {
      const int k = ks * 4;
      v2f a, b0, b1 = {};
      a.x = A[(ti + lm) * LD + k + kh];
      a.y = A[(ti + lm) * LD + k + kh + 1];
      if (TB) {
        b0.x = B[lm * LD + k + kh];
        b0.y = B[lm * LD + k + kh + 1];
      } else {
        b0.x = B[(k + kh    ) * LD + lm];
        b0.y = B[(k + kh + 1) * LD + lm];
      }
      acc0 = __builtin_amdgcn_wmma_f32_16x16x4_f32(false, a, false, b0,
                                                   (short)0, acc0, false, false);
      if (COLS > 1) {
        if (TB) {
          b1.x = B[(16 + lm) * LD + k + kh];
          b1.y = B[(16 + lm) * LD + k + kh + 1];
        } else {
          b1.x = B[(k + kh    ) * LD + 16 + lm];
          b1.y = B[(k + kh + 1) * LD + 16 + lm];
        }
        acc1 = __builtin_amdgcn_wmma_f32_16x16x4_f32(false, a, false, b1,
                                                     (short)0, acc1, false, false);
      }
    }
    const int r0 = ti + hi * 8;                 // C/D layout: M = v + 8*(lane>>4)
#pragma unroll
    for (int vv = 0; vv < 8; ++vv) {
      C[(r0 + vv) * LD + lm] = acc0[vv];
      if (COLS > 1) C[(r0 + vv) * LD + 16 + lm] = acc1[vv];
    }
  }
#elif defined(__gfx1250__)
  // Fallback: probe-confirmed f16 WMMA (16x16x32) — one wmma covers full K=32.
#pragma unroll 1
  for (int ti = 0; ti < ROWS * 16; ti += 16) {
#pragma unroll 1
    for (int tj = 0; tj < COLS * 16; tj += 16) {
      v8f  acc = {};
      v16h a, b;
      const int m = ti + lm;
      const int n = tj + lm;
#pragma unroll
      for (int j = 0; j < 4; ++j) {
        a[2*j]     = (_Float16)A[m*LD +      2*j     + 8*hi];
        a[2*j+1]   = (_Float16)A[m*LD +      2*j + 1 + 8*hi];
        a[8+2*j]   = (_Float16)A[m*LD + 16 + 2*j     + 8*hi];
        a[8+2*j+1] = (_Float16)A[m*LD + 16 + 2*j + 1 + 8*hi];
      }
#pragma unroll
      for (int j = 0; j < 8; ++j) {
        const int k0 = 2*j + 8*hi + ((j >= 4) ? 8 : 0);
        if (TB) {
          b[2*j]   = (_Float16)B[n*LD + k0];
          b[2*j+1] = (_Float16)B[n*LD + k0 + 1];
        } else {
          b[2*j]   = (_Float16)B[(k0    )*LD + n];
          b[2*j+1] = (_Float16)B[(k0 + 1)*LD + n];
        }
      }
      acc = __builtin_amdgcn_wmma_f32_16x16x32_f16(false, a, false, b,
                                                   (short)0, acc, false, false);
      const int r0 = ti + hi * 8;
#pragma unroll
      for (int vv = 0; vv < 8; ++vv)
        C[(r0 + vv) * LD + tj + lm] = acc[vv];
    }
  }
#else
  // Host-compile placeholder (parsed only; device uses the WMMA paths above).
  for (int c = 0; c < COLS * 16; ++c) {
    float s = 0.f;
    for (int k = 0; k < 4 * KS; ++k)
      s += A[lane * LD + k] * (TB ? B[c * LD + k] : B[k * LD + c]);
    C[lane * LD + c] = s;
  }
#endif
}

// =============================== persistent EKF kernel ===============================
// Cap at 256 arch VGPRs: denies the allocator the high-VGPR bank so every WMMA/DS
// region stays free of s_set_vgpr_msb toggles (SALU on the critical issue path);
// hoisted LDS address constants get rematerialized instead.
__global__ __launch_bounds__(32)
__attribute__((amdgpu_num_vgpr(256)))
void iekf_scan_kernel(
    const float* __restrict t, const float* __restrict u, const float* __restrict mcov,
    const float* __restrict v_mes, const float* __restrict ang0,
    const float* __restrict wic, float* __restrict out, int N)
{
  __shared__ float sP [32*LD], sF [32*LD], sF2[32*LD], sF3[32*LD];
  __shared__ float sPhi[32*LD], sG [32*LD], sGq[32*LD], sM [32*LD];
  __shared__ float sT [32*LD], sPn[32*LD], sKH[32*LD], sP2[32*LD];
  __shared__ float sHm[16*LD];              // H as 16xLD matrix (rows 0,1 live)
  __shared__ float sPHm[32*LD];             // P_new * H^T (cols 0,1 live)
  __shared__ float sK0[32], sK1[32], sdx[32];

  const int l = (int)threadIdx.x;

  // ---- per-lane replicated carry state ----
  float Rot[9], Rci[9], v[3], p[3], bo[3], ba[3], tci[3];
  from_rpy(ang0[0], ang0[1], ang0[2], Rot);
#pragma unroll
  for (int j = 0; j < 3; ++j) { v[j] = v_mes[j]; p[j]=0.f; bo[j]=0.f; ba[j]=0.f; tci[j]=0.f; }
#pragma unroll
  for (int j = 0; j < 9; ++j) Rci[j] = 0.f;
  Rci[0] = Rci[4] = Rci[8] = 1.f;

  // ---- one-time zero of every LDS tile. The nonzero STRUCTURE of F, G, Phi is
  //      fixed across iterations, so in-loop fills touch only the (data-dependent)
  //      nonzero entries; every structural zero is written exactly once, here. ----
  for (int c = 0; c < LD; ++c) {
    sF [l*LD+c]=0.f; sF2[l*LD+c]=0.f; sF3[l*LD+c]=0.f; sPhi[l*LD+c]=0.f;
    sG [l*LD+c]=0.f; sGq[l*LD+c]=0.f; sM [l*LD+c]=0.f; sT  [l*LD+c]=0.f;
    sPn[l*LD+c]=0.f; sKH[l*LD+c]=0.f; sP2[l*LD+c]=0.f; sP  [l*LD+c]=0.f;
    sPHm[l*LD+c]=0.f;
    if (l < 16) sHm[l*LD+c]=0.f;
  }
  if (l >= 9 && l < P_DIM) sPhi[l*LD + l] = 1.f;   // Phi rows 9..20 are constant identity

  // ---- P0 (init_covariance) ----
  float beta[6];
#pragma unroll
  for (int k = 0; k < 6; ++k) beta[k] = powf(10.f, tanhf(wic[k]));
  {
    float val = 0.f;
    if      (l < 2)             val = 1e-3f * beta[0];
    else if (l == 3 || l == 4)  val = 0.1f  * beta[1];
    else if (l >= 9  && l < 12) val = 0.006f* beta[2];
    else if (l >= 12 && l < 15) val = 0.004f* beta[3];
    else if (l >= 15 && l < 18) val = 1e-6f * beta[4];
    else if (l >= 18 && l < 21) val = 0.005f* beta[5];
    sP[l*LD + l] = val;                        // diagonal init
  }

  // output offsets (Rot, v, p, b_omega, b_acc, Rot_c_i, t_c_i concatenated flat)
  const long oR = 0, oV = 9L*N, oP = 12L*N, oBO = 15L*N, oBA = 18L*N, oRC = 21L*N, oTC = 30L*N;
  if (l == 0) {
#pragma unroll
    for (int j = 0; j < 9; ++j) { out[oR + j] = Rot[j]; out[oRC + j] = Rci[j]; }
#pragma unroll
    for (int j = 0; j < 3; ++j) {
      out[oV + j] = v[j]; out[oP + j] = 0.f; out[oBO + j] = 0.f;
      out[oBA + j] = 0.f; out[oTC + j] = 0.f;
    }
  }
  __syncthreads();

  // =============================== sequential scan ===============================
  for (int i = 1; i < N; ++i) {
    const float dt  = t[i] - t[i-1];
    float ui[6];
#pragma unroll
    for (int j = 0; j < 6; ++j) ui[j] = u[(long)i*6 + j];
    const float R0m = mcov[(long)i*2 + 0];
    const float R1m = mcov[(long)i*2 + 1];

    // ---- propagate mean (per-lane) ----
    float am[3] = {ui[3]-ba[0], ui[4]-ba[1], ui[5]-ba[2]};
    float acc[3]; m3vec(Rot, am, acc); acc[2] -= GACC;
    float vn[3], pn[3];
#pragma unroll
    for (int j = 0; j < 3; ++j) {
      vn[j] = v[j] + acc[j]*dt;
      pn[j] = p[j] + v[j]*dt + 0.5f*acc[j]*dt*dt;
    }
    float dphi[3] = {(ui[0]-bo[0])*dt, (ui[1]-bo[1])*dt, (ui[2]-bo[2])*dt};
    float dRm[9]; so3exp(dphi, dRm);
    float RotN[9]; m3mul(Rot, dRm, RotN);

    // ---- sparse fill of F*dt, G*dt, (G*dt)*Q (only structurally-nonzero entries) ----
    float skv[9], skp[9], vsr[9], psr[9];
    skew3(v, skv); m3mul(skv, Rot, vsr);
    skew3(p, skp); m3mul(skp, Rot, psr);
    if (l < 3) {                                   // rows 0..2
      float ro[3]; row3(Rot, l, ro);
#pragma unroll
      for (int j = 0; j < 3; ++j) {
        sF [l*LD + 9 + j] = -ro[j]*dt;             // F[0:3,9:12] = -Rot
        sG [l*LD + j]     =  ro[j]*dt;             // G[0:3,0:3]  =  Rot
        sGq[l*LD + j]     =  ro[j]*dt*1e-3f;
      }
    } else if (l < 6) {                            // rows 3..5
      const int r = l - 3;
      float ro[3], vr[3]; row3(Rot, r, ro); row3(vsr, r, vr);
      if (l == 3) sF[l*LD + 1] =  GACC*dt;         // skew(g) entries
      if (l == 4) sF[l*LD + 0] = -GACC*dt;
#pragma unroll
      for (int j = 0; j < 3; ++j) {
        sF [l*LD +  9 + j] = -vr[j]*dt;            // -skew(v)Rot
        sF [l*LD + 12 + j] = -ro[j]*dt;            // -Rot
        sG [l*LD + j]      =  vr[j]*dt;
        sGq[l*LD + j]      =  vr[j]*dt*1e-3f;
        sG [l*LD + 3 + j]  =  ro[j]*dt;
        sGq[l*LD + 3 + j]  =  ro[j]*dt*1e-2f;
      }
    } else if (l < 9) {                            // rows 6..8
      const int r = l - 6;
      float pr[3]; row3(psr, r, pr);
      sF[l*LD + 3 + r] = dt;                       // I3 block
#pragma unroll
      for (int j = 0; j < 3; ++j) {
        sF [l*LD + 9 + j] = -pr[j]*dt;             // -skew(p)Rot
        sG [l*LD + j]     =  pr[j]*dt;
        sGq[l*LD + j]     =  pr[j]*dt*1e-3f;
      }
    } else if (l < P_DIM) {                        // rows 9..20: single diagonal entry
      const float q = (l < 12) ? 6e-9f : ((l < 15) ? 2e-4f : 1e-9f);
      sG [l*LD + (l-3)] = dt;
      sGq[l*LD + (l-3)] = dt*q;
    }
    __syncthreads();

    // F nonzero: rows<9, cols<15  ->  F^2, F^3 live in tile (0,0), inner dim < 12
    mm32<false,1,1,3>(sF,  sF, sF2, l);  __syncthreads();   // F^2
    mm32<false,1,1,3>(sF2, sF, sF3, l);  __syncthreads();   // F^3

    if (l < 9)                                     // Phi rows 0..8 vary, cols < 15
      for (int c = 0; c < 15; ++c)
        sPhi[l*LD + c] = ((l == c) ? 1.f : 0.f)
                       + sF[l*LD + c] + 0.5f*sF2[l*LD + c] + (1.f/6.f)*sF3[l*LD + c];
    __syncthreads();

    mm32<true,2,2,5>(sGq, sG, sM, l);    __syncthreads();    // G Q G^T (inner dim 18)
    if (l < P_DIM)
      for (int c = 0; c < P_DIM; ++c) sM[l*LD + c] += sP[l*LD + c];
    __syncthreads();
    mm32<false,2,2,6>(sPhi, sM, sT, l);  __syncthreads();    // Phi (P + GQG^T)
    mm32<true ,2,2,6>(sT, sPhi, sPn, l);                     // ... Phi^T  -> P_new

    // ---- measurement model (per-lane, register-only; overlaps Pn stores) ----
    float Rb[9];  m3mul(RotN, Rci, Rb);                // Rot_body
    float vi[3];  m3tvec(RotN, vn, vi);                // v_imu
    float om[3] = {ui[0]-bo[0], ui[1]-bo[1], ui[2]-bo[2]};
    float tv[3];  m3tvec(Rci, vi, tv);
    float st[9];  skew3(tci, st);
    float sw[3];  m3vec(st, om, sw);
    const float r0 = -(tv[1] + sw[1]);
    const float r1 = -(tv[2] + sw[2]);
    float svi[9]; skew3(vi, svi);
    float Cv[9];  m3tmulm(Rci, svi, Cv);

    float h0 = 0.f, h1 = 0.f;                          // H column l (branch-free selects)
    if (l >= 3 && l < 6) {
      const int r = l - 3;
      h0 = (r==0) ? Rb[1] : ((r==1) ? Rb[4] : Rb[7]);  // col 1 of Rot_body
      h1 = (r==0) ? Rb[2] : ((r==1) ? Rb[5] : Rb[8]);  // col 2 of Rot_body
    } else if (l >= 9 && l < 12) {
      const int r = l - 9;                             // rows 1,2 of skew(t_c_i)
      h0 = (r==0) ?  tci[2] : ((r==1) ? 0.f     : -tci[0]);
      h1 = (r==0) ? -tci[1] : ((r==1) ? tci[0]  : 0.f);
    } else if (l >= 15 && l < 18) {
      const int r = l - 15;                            // rows 1,2 of Rci^T skew(v_imu)
      h0 = (r==0) ? Cv[3] : ((r==1) ? Cv[4] : Cv[5]);
      h1 = (r==0) ? Cv[6] : ((r==1) ? Cv[7] : Cv[8]);
    } else if (l >= 18 && l < P_DIM) {
      const int r = l - 18;                            // rows 1,2 of -skew(omega)
      h0 = (r==0) ? -om[2] : ((r==1) ? 0.f    : om[0]);
      h1 = (r==0) ?  om[1] : ((r==1) ? -om[0] : 0.f);
    }
    if (l < P_DIM) { sHm[l] = h0; sHm[LD + l] = h1; }
    __syncthreads();                                   // Pn stores + H visible

    mm32<true,2,1,6>(sPn, sHm, sPHm, l); __syncthreads();   // PHt = P_new H^T (cols 0,1)

    const float ph0 = sPHm[l*LD + 0];
    const float ph1 = sPHm[l*LD + 1];

    // S = H P H^T + R via 5-step wave butterfly (all lanes end with full sums)
    float t00 = h0*ph0, t01 = h0*ph1, t10 = h1*ph0, t11 = h1*ph1;
#pragma unroll
    for (int m = 16; m >= 1; m >>= 1) {
      t00 += __shfl_xor(t00, m, 32);
      t01 += __shfl_xor(t01, m, 32);
      t10 += __shfl_xor(t10, m, 32);
      t11 += __shfl_xor(t11, m, 32);
    }
    const float S00 = t00 + R0m, S01 = t01, S10 = t10, S11 = t11 + R1m;
    const float idet = 1.f / (S00*S11 - S01*S10);
    const float si00 =  S11*idet, si01 = -S01*idet, si10 = -S10*idet, si11 =  S00*idet;
    float K0 = ph0*si00 + ph1*si10;
    float K1 = ph0*si01 + ph1*si11;
    if (l >= P_DIM) { K0 = 0.f; K1 = 0.f; }
    sK0[l] = K0; sK1[l] = K1;
    sdx[l] = K0*r0 + K1*r1;
    __syncthreads();

    float dx[21];
#pragma unroll
    for (int j = 0; j < 21; ++j) dx[j] = sdx[j];

    // I - K H (21x21 region; padding stays zero from startup)
    if (l < P_DIM)
      for (int c = 0; c < P_DIM; ++c)
        sKH[l*LD + c] = ((l == c) ? 1.f : 0.f) - (K0*sHm[c] + K1*sHm[LD + c]);
    __syncthreads();

    mm32<false,2,2,6>(sKH, sPn, sT, l);  __syncthreads();    // (I-KH) P
    mm32<true ,2,2,6>(sT, sKH, sP2, l);  __syncthreads();    // ... (I-KH)^T

    if (l < P_DIM)                                            // + K R K^T (rank-2)
      for (int j = 0; j < P_DIM; ++j)
        sP2[l*LD + j] += R0m*K0*sK0[j] + R1m*K1*sK1[j];
    __syncthreads();

    if (l < P_DIM)                                            // symmetrize into carry P
      for (int c = 0; c < P_DIM; ++c)
        sP[l*LD + c] = 0.5f*(sP2[l*LD + c] + sP2[c*LD + l]);
    __syncthreads();

    // ---- state update via SE_2(3) exp ----
    float dR[9], J[9];
    sen3exp_rj(dx, dR, J);
    float RotU[9]; m3mul(dR, RotN, RotU);
    float dv[3], dp[3], vU[3], pU[3];
    m3vec(J, dx+3, dv); m3vec(J, dx+6, dp);
    m3vec(dR, vn, vU);  m3vec(dR, pn, pU);
#pragma unroll
    for (int j = 0; j < 3; ++j) {
      vU[j] += dv[j]; pU[j] += dp[j];
      bo[j] += dx[9 + j]; ba[j] += dx[12 + j]; tci[j] += dx[18 + j];
    }
    float dRc[9]; so3exp(dx + 15, dRc);
    float RciU[9]; m3mul(dRc, Rci, RciU);
#pragma unroll
    for (int j = 0; j < 9; ++j) { Rot[j] = RotU[j]; Rci[j] = RciU[j]; }
#pragma unroll
    for (int j = 0; j < 3; ++j) { v[j] = vU[j]; p[j] = pU[j]; }

    if (l == 0) {
#pragma unroll
      for (int j = 0; j < 9; ++j) { out[oR + 9L*i + j] = Rot[j]; out[oRC + 9L*i + j] = Rci[j]; }
#pragma unroll
      for (int j = 0; j < 3; ++j) {
        out[oV + 3L*i + j] = v[j];  out[oP  + 3L*i + j] = p[j];
        out[oBO + 3L*i + j] = bo[j]; out[oBA + 3L*i + j] = ba[j];
        out[oTC + 3L*i + j] = tci[j];
      }
    }
    __syncthreads();
  }
}

extern "C" void kernel_launch(void* const* d_in, const int* in_sizes, int n_in,
                              void* d_out, int out_size, void* d_ws, size_t ws_size,
                              hipStream_t stream) {
  (void)n_in; (void)out_size; (void)d_ws; (void)ws_size;
  const float* t    = (const float*)d_in[0];
  const float* u    = (const float*)d_in[1];
  const float* mcov = (const float*)d_in[2];
  const float* vmes = (const float*)d_in[3];
  // d_in[4] = p_mes (unused by the reference computation)
  const float* ang0 = (const float*)d_in[5];
  const float* wic  = (const float*)d_in[6];
  const int N = in_sizes[0];
  iekf_scan_kernel<<<dim3(1), dim3(32), 0, stream>>>(t, u, mcov, vmes, ang0, wic,
                                                     (float*)d_out, N);
}
